// BahdanauAttention_44719199486043
// MI455X (gfx1250) — matrix operand order repaired and checked by reading, not compiled, not run on hardware
//
#include <hip/hip_runtime.h>
#include <hip/hip_bf16.h>
#include <cstdint>
#include <cstddef>

#define NB   64
#define NT   2048
#define NE   512
#define NA   256

typedef __attribute__((ext_vector_type(16))) __bf16 v16bf;
typedef __attribute__((ext_vector_type(8)))  __bf16 v8bf;
typedef __attribute__((ext_vector_type(4)))  __bf16 v4bf;
typedef __attribute__((ext_vector_type(8)))  float  v8f;

union Frag { v16bf v; v8bf h[2]; };

__device__ __forceinline__ float fast_tanh(float x) {
  float y;
  asm volatile("v_tanh_f32 %0, %1\n\tv_nop" : "=v"(y) : "v"(x));
  return y;
}

__device__ __forceinline__ void async_copy64(uint32_t lds_off,
                                             unsigned long long gaddr) {
  asm volatile(
      "global_load_async_to_lds_b128 %0, %1, off\n\t"
      "global_load_async_to_lds_b128 %0, %1, off offset:16\n\t"
      "global_load_async_to_lds_b128 %0, %1, off offset:32\n\t"
      "global_load_async_to_lds_b128 %0, %1, off offset:48"
      :: "v"(lds_off), "v"(gaddr) : "memory");
}
__device__ __forceinline__ void wait_async0() {
  asm volatile("s_wait_asynccnt 0x0" ::: "memory");
}

__global__ void __launch_bounds__(256)
pack_wenc_kernel(const float* __restrict__ W, __bf16* __restrict__ Wt) {
  __shared__ float tile[32][33];
  const int k0 = blockIdx.x * 32, n0 = blockIdx.y * 32;
  const int tx = threadIdx.x & 31, ty = threadIdx.x >> 5;
  #pragma unroll
  for (int i = 0; i < 32; i += 8)
    tile[ty + i][tx] = W[(size_t)(k0 + ty + i) * NA + n0 + tx];
  __syncthreads();
  #pragma unroll
  for (int i = 0; i < 32; i += 8)
    Wt[(size_t)(n0 + ty + i) * NE + k0 + tx] = (__bf16)tile[tx][ty + i];
}

__global__ void __launch_bounds__(256)
proj_dec_kernel(const float* __restrict__ dec, const float* __restrict__ Wdec,
                float* __restrict__ pd) {
  __shared__ float dh[NE];
  const int b = blockIdx.x, a = threadIdx.x;
  dh[a]       = dec[b * NE + a];
  dh[a + 256] = dec[b * NE + 256 + a];
  __syncthreads();
  float s = 0.f;
  #pragma unroll 8
  for (int k = 0; k < NE; ++k) s = fmaf(dh[k], Wdec[k * NA + a], s);
  pd[b * NA + a] = s;
}

__global__ void __launch_bounds__(256)
energy_kernel(const float* __restrict__ enc, const __bf16* __restrict__ Wt,
              const float* __restrict__ pd, const float* __restrict__ vvec,
              const int* __restrict__ mask, float* __restrict__ energy) {
  constexpr int TM = 64, KT = 32, LSTR = 48;
  __shared__ __align__(16) __bf16 ldsA[TM * LSTR];
  __shared__ __align__(16) __bf16 ldsB[NA * LSTR];
  __shared__ float ered[2 * TM];

  const int tid  = threadIdx.x;
  const int lane = tid & 31, wave = tid >> 5;
  const int wm = wave & 3, wn = wave >> 2;
  const int Nl = lane & 15, hh = lane >> 4;
  const int m0 = blockIdx.x * TM;
  const int b  = m0 >> 11;
  const int t0 = m0 & (NT - 1);

  v8f acc[8] = {};

  const float* encBase = enc + (size_t)m0 * NE;
  const uint32_t ldsBrow = (uint32_t)(uintptr_t)&ldsB[tid * LSTR];
  const unsigned long long wtRow =
      (unsigned long long)(uintptr_t)(Wt + (size_t)tid * NE);
  const int pfRow = tid >> 2;
  const int pfCol = (tid & 3) * 8;

  for (int k0 = 0; k0 < NE; k0 += KT) {
    __syncthreads();
    async_copy64(ldsBrow, wtRow + (size_t)k0 * sizeof(__bf16));

    const int kpf = (k0 + KT < NE) ? (k0 + KT) : k0;
    __builtin_prefetch(&encBase[(size_t)pfRow * NE + kpf + pfCol], 0, 3);

    #pragma unroll
    for (int i = 0; i < 2; ++i) {
      const int idx = tid + i * 256;
      const int r = idx >> 3, g = (idx & 7) * 4;
      const float4 x = *(const float4*)&encBase[(size_t)r * NE + k0 + g];
      v4bf p;
      p[0] = (__bf16)x.x; p[1] = (__bf16)x.y;
      p[2] = (__bf16)x.z; p[3] = (__bf16)x.w;
      *(v4bf*)&ldsA[r * LSTR + g] = p;
    }
    wait_async0();
    __syncthreads();

    Frag a;
    const int row = (wm << 4) + Nl;
    a.h[0] = *(const v8bf*)&ldsA[row * LSTR + hh * 8];
    a.h[1] = *(const v8bf*)&ldsA[row * LSTR + 16 + hh * 8];

    #pragma unroll
    for (int nt = 0; nt < 8; ++nt) {
      const int col = (wn << 7) + (nt << 4) + Nl;
      Frag bf;
      bf.h[0] = *(const v8bf*)&ldsB[col * LSTR + hh * 8];
      bf.h[1] = *(const v8bf*)&ldsB[col * LSTR + 16 + hh * 8];
      acc[nt] = __builtin_amdgcn_wmma_f32_16x16x32_bf16(
          false, a.v, false, bf.v, (short)0, acc[nt], false, false);
    }
  }

  float s[8];
  #pragma unroll
  for (int j = 0; j < 8; ++j) s[j] = 0.f;
  const float* pdb = pd + b * NA;
  #pragma unroll
  for (int nt = 0; nt < 8; ++nt) {
    const int N = (wn << 7) + (nt << 4) + Nl;
    const float pv = pdb[N];
    const float vv = vvec[N];
    #pragma unroll
    for (int j = 0; j < 8; ++j) s[j] += vv * fast_tanh(acc[nt][j] + pv);
  }
  #pragma unroll
  for (int j = 0; j < 8; ++j) {
    float x = s[j];
    #pragma unroll
    for (int off = 1; off < 16; off <<= 1) x += __shfl_xor(x, off, 32);
    if (Nl == 0) ered[wn * TM + (wm << 4) + (hh << 3) + j] = x;
  }
  __syncthreads();
  if (tid < TM) {
    float e = ered[tid] + ered[TM + tid];
    const int t = t0 + tid;
    if (mask[b * NT + t] != 0) e = -__builtin_inff();
    energy[(size_t)b * NT + t] = e;
  }
}

__global__ void __launch_bounds__(256)
softmax_kernel(float* __restrict__ attn) {
  __shared__ float red[8];
  const int b = blockIdx.x, tid = threadIdx.x;
  float* row = attn + (size_t)b * NT;
  float e[8];
  #pragma unroll
  for (int i = 0; i < 8; ++i) e[i] = row[tid + 256 * i];

  float mx = -__builtin_inff();
  #pragma unroll
  for (int i = 0; i < 8; ++i) mx = fmaxf(mx, e[i]);
  #pragma unroll
  for (int off = 16; off > 0; off >>= 1) mx = fmaxf(mx, __shfl_xor(mx, off, 32));
  if ((tid & 31) == 0) red[tid >> 5] = mx;
  __syncthreads();
  mx = red[0];
  #pragma unroll
  for (int i = 1; i < 8; ++i) mx = fmaxf(mx, red[i]);
  __syncthreads();

  const bool allmask = !(mx > -__builtin_inff());
  float w[8];
  float sum = 0.f;
  #pragma unroll
  for (int i = 0; i < 8; ++i) {
    w[i] = allmask ? 0.f : __expf(e[i] - mx);
    sum += w[i];
  }
  #pragma unroll
  for (int off = 16; off > 0; off >>= 1) sum += __shfl_xor(sum, off, 32);
  if ((tid & 31) == 0) red[tid >> 5] = sum;
  __syncthreads();
  sum = 0.f;
  #pragma unroll
  for (int i = 0; i < 8; ++i) sum += red[i];
  const float inv = (sum > 0.f) ? (1.f / sum) : 0.f;
  #pragma unroll
  for (int i = 0; i < 8; ++i) row[tid + 256 * i] = w[i] * inv;
}

__global__ void zero_kernel(float* __restrict__ p, int n) {
  const int i = blockIdx.x * blockDim.x + threadIdx.x;
  if (i < n) p[i] = 0.f;
}

__global__ void __launch_bounds__(256)
context_kernel(const float* __restrict__ enc, const float* __restrict__ attn,
               float* __restrict__ ctx) {
  __shared__ float w[256];
  const int b = blockIdx.x, chunk = blockIdx.y, tid = threadIdx.x;
  const int tbase = chunk * 256;
  w[tid] = attn[(size_t)b * NT + tbase + tid];
  __syncthreads();
  const float2* rowb = (const float2*)(enc + ((size_t)b * NT + tbase) * NE);
  float ax = 0.f, ay = 0.f;
  for (int t = 0; t < 256; ++t) {
    const float wt = w[t];
    if (wt != 0.f) {
      const float2 x = rowb[(size_t)t * (NE / 2) + tid];
      ax = fmaf(wt, x.x, ax);
      ay = fmaf(wt, x.y, ay);
    }
  }
  atomicAdd(&ctx[b * NE + 2 * tid],     ax);
  atomicAdd(&ctx[b * NE + 2 * tid + 1], ay);
}

extern "C" void kernel_launch(void* const* d_in, const int* in_sizes, int n_in,
                              void* d_out, int out_size, void* d_ws, size_t ws_size,
                              hipStream_t stream) {
  const float* enc  = (const float*)d_in[0];
  const float* dec  = (const float*)d_in[1];
  const int*   mask = (const int*)  d_in[2];
  const float* Wenc = (const float*)d_in[3];
  const float* Wdec = (const float*)d_in[4];
  const float* v    = (const float*)d_in[5];

  float* ctx  = (float*)d_out;
  float* attn = (float*)d_out + NB * NE;
  float*  pd  = (float*)d_ws;
  __bf16* Wt  = (__bf16*)((char*)d_ws + (size_t)NB * NA * sizeof(float));

  pack_wenc_kernel<<<dim3(NE / 32, NA / 32), 256, 0, stream>>>(Wenc, Wt);
  proj_dec_kernel<<<NB, 256, 0, stream>>>(dec, Wdec, pd);
  energy_kernel<<<(NB * NT) / 64, 256, 0, stream>>>(enc, Wt, pd, v, mask, attn);
  softmax_kernel<<<NB, 256, 0, stream>>>(attn);
  zero_kernel<<<(NB * NE + 255) / 256, 256, 0, stream>>>(ctx, NB * NE);
  context_kernel<<<dim3(NB, NT / 256), 256, 0, stream>>>(enc, attn, ctx);
}
